// MultiheadAttention_23356032155930
// MI455X (gfx1250) — compile-verified, matrix operand order checked by reading, not run on hardware
//
#include <hip/hip_runtime.h>
#include <hip/hip_bf16.h>
#include <cstdint>

#define B_  2
#define T_  2048
#define E_  1024
#define H_  16
#define D_  64
#define BH_ (B_ * H_)
#define NEGV (-1e9f)

#define AS1 __attribute__((address_space(1)))
#define AS3 __attribute__((address_space(3)))

typedef __attribute__((ext_vector_type(16))) _Float16 v16h;
typedef __attribute__((ext_vector_type(8)))  _Float16 v8h;
typedef __attribute__((ext_vector_type(8)))  float    v8f;
typedef int i32x4 __attribute__((vector_size(16)));

#define WMMA_F32_F16(a, b, c) \
  __builtin_amdgcn_wmma_f32_16x16x32_f16(false, (a), false, (b), (short)0, (c), false, false)

// Load a 16x32 f16 A-style fragment from row-major [16, ld] data.
// ISA 7.12.2: lane L holds row L%16; element e holds K = e%8 + (e/8)*16 + (L/16)*8,
// i.e. two contiguous 8-half (16B) chunks per lane.
__device__ __forceinline__ v16h load_frag(const _Float16* base, int ld) {
  const int lane = threadIdx.x & 31;
  const int row  = lane & 15;
  const int koff = (lane >> 4) * 8;
  const _Float16* p = base + (size_t)row * ld + koff;
  v8h lo = *(const v8h*)(p);
  v8h hi = *(const v8h*)(p + 16);
  return __builtin_shufflevector(lo, hi, 0,1,2,3,4,5,6,7,8,9,10,11,12,13,14,15);
}

__global__ void cvt_f32_f16(const float* __restrict__ s, _Float16* __restrict__ d, int n) {
  for (int i = blockIdx.x * blockDim.x + threadIdx.x; i < n; i += gridDim.x * blockDim.x)
    d[i] = (_Float16)s[i];
}

// y = X @ W^T + bias, fused with head split + (q scale) + xpos rotary.
// Each wave computes 4 adjacent 16-col tiles, reusing one A fragment per K-chunk
// across 4 WMMAs (raises WMMA density; A traffic /4).
// mode 0: q -> dst[BH,T,D] (scale 1/8, xpos upscale)
// mode 1: k -> dst[BH,T,D] (xpos downscale)
// mode 2: v -> dst[BH,D,T] (transposed for PV B-fragments)
__global__ __launch_bounds__(128) void proj_qkv_kernel(
    const _Float16* __restrict__ X, const _Float16* __restrict__ W,
    const float* __restrict__ bias, _Float16* __restrict__ dst, int mode) {
  const int wave = threadIdx.x >> 5, lane = threadIdx.x & 31;
  const int m0 = blockIdx.x * 16;
  const int n0 = blockIdx.y * 256 + wave * 64;   // 4 x 16 columns per wave
  v8f c0 = {}, c1 = {}, c2 = {}, c3 = {};
  for (int k0 = 0; k0 < E_; k0 += 32) {
    v16h a  = load_frag(X + (size_t)m0 * E_ + k0, E_);
    v16h b0 = load_frag(W + (size_t)(n0 +  0) * E_ + k0, E_);
    v16h b1 = load_frag(W + (size_t)(n0 + 16) * E_ + k0, E_);
    v16h b2 = load_frag(W + (size_t)(n0 + 32) * E_ + k0, E_);
    v16h b3 = load_frag(W + (size_t)(n0 + 48) * E_ + k0, E_);
    c0 = WMMA_F32_F16(a, b0, c0);
    c1 = WMMA_F32_F16(a, b1, c1);
    c2 = WMMA_F32_F16(a, b2, c2);
    c3 = WMMA_F32_F16(a, b3, c3);
  }
  v8f acc[4] = {c0, c1, c2, c3};
#pragma unroll
  for (int u = 0; u < 4; ++u) {
    const int n = n0 + u * 16 + (lane & 15);
    const float bn = bias[n];
    const int d = n & 63, hd = n >> 6, j = d >> 1;
    const float sgn  = (mode == 1) ? -1.f : 1.f;        // downscale for k
    const float logb = __logf((2.f * (float)j + 0.4f * (float)D_) / (1.4f * (float)D_));
    const float invf = __expf(-(float)j * (9.210340371976184f / (float)(D_ / 2)));
#pragma unroll
    for (int r = 0; r < 8; ++r) {
      float v = acc[u][r] + bn;
      if (mode == 0) v *= 0.125f;                       // D^-0.5
      const int m = m0 + r + ((lane >> 4) << 3);
      const int b_ = m / T_, t = m % T_;
      const int bh = b_ * H_ + hd;
      if (mode < 2) {
        // rotate_every_two: partner element sits in lane^1 (adjacent column)
        float p   = __shfl_xor(v, 1, 32);
        float rot = (n & 1) ? p : -p;
        float pos = (float)t - (float)(T_ / 2);
        float sc  = __expf(logb * pos * (sgn / 512.0f));
        float ang = (float)t * invf;
        v = v * __cosf(ang) * sc + rot * __sinf(ang) * sc;
        dst[((size_t)bh * T_ + t) * D_ + d] = (_Float16)v;
      } else {
        dst[((size_t)bh * D_ + d) * T_ + t] = (_Float16)v;
      }
    }
  }
}

// One block = one (bh, 16-row tile): scores -> softmax -> weights out -> P@V.
__global__ __launch_bounds__(256) void attn_kernel(
    const _Float16* __restrict__ qh, const _Float16* __restrict__ kh,
    const _Float16* __restrict__ vT, const float* __restrict__ attn_mask,
    const unsigned char* __restrict__ kpm, const float* __restrict__ rel_pos,
    float* __restrict__ w_out, _Float16* __restrict__ ctx_h) {
  extern __shared__ char smem[];
  float*    S  = (float*)smem;                      // [16][T_] fp32  (128 KB)
  _Float16* P  = (_Float16*)(smem + 16 * T_ * 4);   // [16][T_] f16   (64 KB)
  _Float16* Qs = P + 16 * T_;                       // [16][D_] f16   (2 KB)
  const int bh = blockIdx.x, t0 = blockIdx.y * 16;
  const int b_ = bh / H_, hh = bh % H_;
  const int wave = threadIdx.x >> 5, lane = threadIdx.x & 31;
  const int tid = threadIdx.x;

  // Stage Q tile (16x64 f16) into LDS once per block via async copy (ASYNCcnt path).
  const _Float16* qbase = qh + ((size_t)bh * T_ + t0) * D_;
#if __has_builtin(__builtin_amdgcn_global_load_async_to_lds_b128)
  if (tid < 128) {
    const _Float16* g = qbase + tid * 8;            // 16B per lane
    __builtin_amdgcn_global_load_async_to_lds_b128(
        (AS1 i32x4*)g, (AS3 i32x4*)(Qs + tid * 8), 0, 0);
  }
#if __has_builtin(__builtin_amdgcn_s_wait_asynccnt)
  __builtin_amdgcn_s_wait_asynccnt(0);
#else
  asm volatile("s_wait_asynccnt 0x0" ::: "memory");
#endif
#else
  if (tid < 128) *(v8h*)(Qs + tid * 8) = *(const v8h*)(qbase + tid * 8);
#endif
  __syncthreads();

  v16h qa0 = load_frag(Qs, D_);
  v16h qa1 = load_frag(Qs + 32, D_);

  // Phase 1: S = q k^T + masks + rel_pos (8 waves x 16 column tiles each)
  for (int ct = wave; ct < T_ / 16; ct += 8) {
    const int s0 = ct * 16;
    if (ct + 8 < T_ / 16) {                         // prefetch next tile's rel_pos rows
      const float* rp = rel_pos + ((size_t)bh * T_ + t0 + (lane >> 1)) * T_ + (s0 + 128);
      __builtin_prefetch(rp, 0, 1);
    }
    const _Float16* kb = kh + ((size_t)bh * T_ + s0) * D_;
    v16h b0 = load_frag(kb, D_);
    v16h b1 = load_frag(kb + 32, D_);
    v8f c = {};
    c = WMMA_F32_F16(qa0, b0, c);
    c = WMMA_F32_F16(qa1, b1, c);
    const int s = s0 + (lane & 15);
    const bool pad = kpm[b_ * T_ + s] != 0;
#pragma unroll
    for (int r = 0; r < 8; ++r) {
      const int m = r + ((lane >> 4) << 3);
      const int t = t0 + m;
      float v = c[r] + attn_mask[(size_t)t * T_ + s];
      if (pad) v = NEGV;
      v += rel_pos[((size_t)bh * T_ + t) * T_ + s];
      S[m * T_ + s] = v;
    }
  }
  __syncthreads();

  // Phase 2: exact row softmax; write fp32 weights once to HBM, f16 P to LDS.
  {
    const int row = wave * 2 + (lane >> 4);
    const int c0  = lane & 15;
    float mx = -3.4e38f;
    for (int i = 0; i < T_ / 16; ++i) mx = fmaxf(mx, S[row * T_ + c0 + i * 16]);
    for (int off = 1; off < 16; off <<= 1) mx = fmaxf(mx, __shfl_xor(mx, off, 32));
    float sum = 0.f;
    for (int i = 0; i < T_ / 16; ++i) {
      const int idx = row * T_ + c0 + i * 16;
      const float e = __expf(S[idx] - mx);
      S[idx] = e;
      sum += e;
    }
    for (int off = 1; off < 16; off <<= 1) sum += __shfl_xor(sum, off, 32);
    const float rinv = 1.f / sum;
    const size_t wbase = (((size_t)hh * B_ + b_) * T_ + (t0 + row)) * (size_t)T_;
    for (int i = 0; i < T_ / 16; ++i) {
      const int cc = c0 + i * 16;
      const float p = S[row * T_ + cc] * rinv;
      w_out[wbase + cc] = p;                        // attn_weights [H,B,T,T]
      P[row * T_ + cc] = (_Float16)p;
    }
  }
  __syncthreads();

  // Phase 3: ctx = P @ V. 8 waves = 4 d-tiles x 2 K-halves; combine via LDS.
  const int dt = wave >> 1, half = wave & 1;
  v8f c = {};
  const int sbeg = half * (T_ / 2);
  for (int s0 = sbeg; s0 < sbeg + T_ / 2; s0 += 32) {
    v16h pa = load_frag(P + s0, T_);
    v16h vb = load_frag(vT + ((size_t)bh * D_ + dt * 16) * T_ + s0, T_);
    c = WMMA_F32_F16(pa, vb, c);
  }
  float* scratch = S;                               // S no longer needed
  if (half == 1)
    for (int r = 0; r < 8; ++r) scratch[dt * 256 + lane * 8 + r] = c[r];
  __syncthreads();
  if (half == 0) {
    for (int r = 0; r < 8; ++r) c[r] += scratch[dt * 256 + lane * 8 + r];
    const int dcol = dt * 16 + (lane & 15);
    for (int r = 0; r < 8; ++r) {
      const int t = t0 + r + ((lane >> 4) << 3);
      ctx_h[(size_t)(b_ * T_ + t) * E_ + hh * D_ + dcol] = (_Float16)c[r];
    }
  }
}

// out = ctx @ wo^T + bo  (fp32 result straight to d_out), same 4x N-tiling.
__global__ __launch_bounds__(128) void out_proj_kernel(
    const _Float16* __restrict__ X, const _Float16* __restrict__ W,
    const float* __restrict__ bias, float* __restrict__ out) {
  const int wave = threadIdx.x >> 5, lane = threadIdx.x & 31;
  const int m0 = blockIdx.x * 16;
  const int n0 = blockIdx.y * 256 + wave * 64;
  v8f c0 = {}, c1 = {}, c2 = {}, c3 = {};
  for (int k0 = 0; k0 < E_; k0 += 32) {
    v16h a  = load_frag(X + (size_t)m0 * E_ + k0, E_);
    v16h b0 = load_frag(W + (size_t)(n0 +  0) * E_ + k0, E_);
    v16h b1 = load_frag(W + (size_t)(n0 + 16) * E_ + k0, E_);
    v16h b2 = load_frag(W + (size_t)(n0 + 32) * E_ + k0, E_);
    v16h b3 = load_frag(W + (size_t)(n0 + 48) * E_ + k0, E_);
    c0 = WMMA_F32_F16(a, b0, c0);
    c1 = WMMA_F32_F16(a, b1, c1);
    c2 = WMMA_F32_F16(a, b2, c2);
    c3 = WMMA_F32_F16(a, b3, c3);
  }
  v8f acc[4] = {c0, c1, c2, c3};
#pragma unroll
  for (int u = 0; u < 4; ++u) {
    const int n = n0 + u * 16 + (lane & 15);
    const float bn = bias[n];
#pragma unroll
    for (int r = 0; r < 8; ++r) {
      const int m = m0 + r + ((lane >> 4) << 3);
      out[(size_t)m * E_ + n] = acc[u][r] + bn;
    }
  }
}

extern "C" void kernel_launch(void* const* d_in, const int* in_sizes, int n_in,
                              void* d_out, int out_size, void* d_ws, size_t ws_size,
                              hipStream_t stream) {
  (void)in_sizes; (void)n_in; (void)out_size; (void)ws_size;
  const float* query = (const float*)d_in[0];
  const float* key   = (const float*)d_in[1];
  const float* value = (const float*)d_in[2];
  const float* wq = (const float*)d_in[3];
  const float* bq = (const float*)d_in[4];
  const float* wk = (const float*)d_in[5];
  const float* bk = (const float*)d_in[6];
  const float* wv = (const float*)d_in[7];
  const float* bv = (const float*)d_in[8];
  const float* wo = (const float*)d_in[9];
  const float* bo = (const float*)d_in[10];
  const float* attn_mask = (const float*)d_in[11];
  const unsigned char* kpm = (const unsigned char*)d_in[12];
  const float* rel_pos = (const float*)d_in[13];

  const size_t BTE = (size_t)B_ * T_ * E_;   // 4,194,304
  const size_t EE  = (size_t)E_ * E_;        // 1,048,576
  _Float16* ws   = (_Float16*)d_ws;          // total 64 MiB of f16 scratch
  _Float16* xq   = ws;
  _Float16* xk   = xq  + BTE;
  _Float16* xv   = xk  + BTE;
  _Float16* wqh  = xv  + BTE;
  _Float16* wkh  = wqh + EE;
  _Float16* wvh  = wkh + EE;
  _Float16* woh  = wvh + EE;
  _Float16* qhb  = woh + EE;                 // [BH,T,D]
  _Float16* khb  = qhb + BTE;                // [BH,T,D]
  _Float16* vTb  = khb + BTE;                // [BH,D,T]
  _Float16* ctxh = vTb + BTE;                // [B,T,E]

  float* out_f = (float*)d_out;              // [B,T,E]
  float* w_out = out_f + BTE;                // [H,B,T,T]

  cvt_f32_f16<<<2048, 256, 0, stream>>>(query, xq, (int)BTE);
  cvt_f32_f16<<<2048, 256, 0, stream>>>(key,   xk, (int)BTE);
  cvt_f32_f16<<<2048, 256, 0, stream>>>(value, xv, (int)BTE);
  cvt_f32_f16<<<1024, 256, 0, stream>>>(wq, wqh, (int)EE);
  cvt_f32_f16<<<1024, 256, 0, stream>>>(wk, wkh, (int)EE);
  cvt_f32_f16<<<1024, 256, 0, stream>>>(wv, wvh, (int)EE);
  cvt_f32_f16<<<1024, 256, 0, stream>>>(wo, woh, (int)EE);

  dim3 pg(B_ * T_ / 16, E_ / 256);
  proj_qkv_kernel<<<pg, 128, 0, stream>>>(xq, wqh, bq, qhb, 0);
  proj_qkv_kernel<<<pg, 128, 0, stream>>>(xk, wkh, bk, khb, 1);
  proj_qkv_kernel<<<pg, 128, 0, stream>>>(xv, wvh, bv, vTb, 2);

  dim3 ag(BH_, T_ / 16);
  size_t lds = (size_t)16 * T_ * 4 + (size_t)16 * T_ * 2 + (size_t)16 * D_ * 2;  // 194 KB
  attn_kernel<<<ag, 256, lds, stream>>>(qhb, khb, vTb, attn_mask, kpm, rel_pos, w_out, ctxh);

  out_proj_kernel<<<pg, 128, 0, stream>>>(ctxh, woh, bo, out_f);
}
